// MahalanobisLayer_50448685859469
// MI455X (gfx1250) — matrix operand order repaired and checked by reading, not compiled, not run on hardware
//
#include <hip/hip_runtime.h>
#include <hip/hip_bf16.h>
#include <stdint.h>

// CDNA5 / gfx1250: wave32, WMMA 16x16x32 bf16 path.
typedef __attribute__((ext_vector_type(16))) __bf16 v16bf;
typedef __attribute__((ext_vector_type(8)))  float  v8f;

#define DIM 512

// ---- bf16 split helpers (round-to-nearest-even) ----
__device__ __forceinline__ uint16_t f32_to_bf16(float f) {
  uint32_t u = __float_as_uint(f);
  u += 0x7FFFu + ((u >> 16) & 1u);
  return (uint16_t)(u >> 16);
}
__device__ __forceinline__ float bf16_to_f32(uint16_t h) {
  return __uint_as_float(((uint32_t)h) << 16);
}

// A-operand (16-bit A 16x32, ISA 7.12.2) slot within one K-chunk region:
//   lane = (r&15) + 16*((kk>>3)&1)
//   v    = (kk&16) ? 4 + ((kk&7)>>1) : ((kk&7)>>1) ;  half = 2v + (kk&1)
__device__ __forceinline__ int slotA(int kc, int kk, int r) {
  int lane = (r & 15) + (((kk >> 3) & 1) << 4);
  int v = (kk & 16) ? (4 + ((kk & 7) >> 1)) : ((kk & 7) >> 1);
  int h = 2 * v + (kk & 1);
  return ((kc * 32 + lane) << 4) + h;
}

// Per-wave LDS block layout (contiguous 64 KB so all DS offsets fit the
// 16-bit immediate): [a0h | a0l | a1h | a1l], 8192 halves (16 KB) each.
#define REG_A0H 0
#define REG_A0L 8192
#define REG_A1H 16384
#define REG_A1L 24576

__device__ __forceinline__ float lds_delta(const uint16_t* aw, int hiReg, int loReg,
                                           int r, int c) {
  int s = slotA(c >> 5, c & 31, r);
  return bf16_to_f32(aw[hiReg + s]) + bf16_to_f32(aw[loReg + s]);
}

// ---- Kernel 1: split S_inv into bf16 hi/lo, scattered into B-operand layout.
// B (16-bit 32x16): lane = N + 16*((kk>>3)&1), half = (kk&7) + 8*(kk>>4) (matches A layout).
__global__ __launch_bounds__(512) void MahalanobisLayer_prepS(
    const float* __restrict__ S, uint16_t* __restrict__ SBhi,
    uint16_t* __restrict__ SBlo) {
  int idx = blockIdx.x * 512 + threadIdx.x;   // 0 .. 512*512-1
  int k = idx >> 9, e = idx & 511;
  float s = S[idx];
  uint16_t hi = f32_to_bf16(s);
  uint16_t lo = f32_to_bf16(s - bf16_to_f32(hi));
  int eblk = e >> 4, N = e & 15, kc = k >> 5, kk = k & 31;
  int lane = N + (((kk >> 3) & 1) << 4);
  int slot = (((eblk * 16 + kc) * 32 + lane) << 4) + ((kk & 7) + ((kk >> 4) << 3));
  SBhi[slot] = hi;
  SBlo[slot] = lo;
}

// ---- Kernel 2: fused delta / GEMM / rowwise quadratic form.
//      2 waves/WG; each wave owns TWO 16-row tiles (32 rows) so every
//      B operand from L2 feeds 2 WMMAs (4 concurrent accumulator chains).
__global__ __launch_bounds__(64) void MahalanobisLayer_main(
    const float* __restrict__ x, const float* __restrict__ xf,
    const uint16_t* __restrict__ SBhi, const uint16_t* __restrict__ SBlo,
    float* __restrict__ out) {
  // 64 KB per wave, 128 KB per WG.
  __shared__ uint16_t A_s[2][4 * 8192];

  const int lane = threadIdx.x & 31;
  const int wave = threadIdx.x >> 5;
  const int rowBase = (blockIdx.x * 2 + wave) * 32;

  uint16_t* aw = A_s[wave];

  // ---- Phase 1: delta = x - x_fit; bf16 hi/lo split; store pre-swizzled
  //      into the WMMA A-operand layout (packed 4 halves -> ds_store_b64).
  for (int t = 0; t < 2; ++t) {
    uint16_t* ah = aw + (t ? REG_A1H : REG_A0H);
    uint16_t* al = aw + (t ? REG_A1L : REG_A0L);
    for (int rr = 0; rr < 16; ++rr) {
      const size_t rowOff = (size_t)(rowBase + t * 16 + rr) * DIM;
      const float4* px = reinterpret_cast<const float4*>(x + rowOff);
      const float4* pf = reinterpret_cast<const float4*>(xf + rowOff);
#pragma unroll
      for (int ch = 0; ch < 4; ++ch) {
        int c4 = ch * 32 + lane;       // float4 index; col = 4*c4
        float4 a = px[c4];
        float4 b = pf[c4];
        float d0 = a.x - b.x, d1 = a.y - b.y, d2 = a.z - b.z, d3 = a.w - b.w;
        uint16_t h0 = f32_to_bf16(d0), h1 = f32_to_bf16(d1);
        uint16_t h2 = f32_to_bf16(d2), h3 = f32_to_bf16(d3);
        uint16_t l0 = f32_to_bf16(d0 - bf16_to_f32(h0));
        uint16_t l1 = f32_to_bf16(d1 - bf16_to_f32(h1));
        uint16_t l2 = f32_to_bf16(d2 - bf16_to_f32(h2));
        uint16_t l3 = f32_to_bf16(d3 - bf16_to_f32(h3));
        int col = c4 << 2;
        int s = slotA(col >> 5, col & 31, rr);   // h is 4-aligned here
        uint2 ph, pl;
        ph.x = (uint32_t)h0 | ((uint32_t)h1 << 16);
        ph.y = (uint32_t)h2 | ((uint32_t)h3 << 16);
        pl.x = (uint32_t)l0 | ((uint32_t)l1 << 16);
        pl.y = (uint32_t)l2 | ((uint32_t)l3 << 16);
        *reinterpret_cast<uint2*>(&ah[s]) = ph;
        *reinterpret_cast<uint2*>(&al[s]) = pl;
      }
    }
  }
  __syncthreads();

  // ---- Phase 2: per e-block pair, C = Delta @ S(:,16) via bf16 WMMA with
  //      hi/lo compensation; 4 independent accumulator chains.
  // All A loads: one base VGPR (aw + lane*32B) + immediate region/kc offsets.
  const v16bf* awV  = reinterpret_cast<const v16bf*>(aw) + lane;   // stride 32 v16bf per kc
  const v16bf* bhiV = reinterpret_cast<const v16bf*>(SBhi) + lane;
  const v16bf* bloV = reinterpret_cast<const v16bf*>(SBlo) + lane;

  float acc0[8], acc1[8];
#pragma unroll
  for (int v = 0; v < 8; ++v) { acc0[v] = 0.0f; acc1[v] = 0.0f; }

  const int n16 = lane & 15;
  const int mhalf = lane >> 4;  // 0: rows 0-7 in C VGPRs, 1: rows 8-15

  for (int ebp = 0; ebp < 16; ++ebp) {
    const int eb0 = ebp * 2, eb1 = eb0 + 1;
    v8f c00 = {}, c01 = {};   // tile0 x {eb0, eb1}
    v8f c10 = {}, c11 = {};   // tile1 x {eb0, eb1}
#pragma unroll 4
    for (int kc = 0; kc < 16; ++kc) {
      // LDS: all offsets < 64 KB from the per-wave base -> DS immediates.
      v16bf a0h = awV[(REG_A0H / 16) + kc * 32];      // 2x ds_load_b128 each
      v16bf a0l = awV[(REG_A0L / 16) + kc * 32];
      v16bf a1h = awV[(REG_A1H / 16) + kc * 32];
      v16bf a1l = awV[(REG_A1L / 16) + kc * 32];
      // Global: base + 24-bit immediate offsets.
      v16bf b0h = bhiV[(eb0 * 16 + kc) * 32];         // 2x global_load_b128 each
      v16bf b0l = bloV[(eb0 * 16 + kc) * 32];
      v16bf b1h = bhiV[(eb1 * 16 + kc) * 32];
      v16bf b1l = bloV[(eb1 * 16 + kc) * 32];
      // 12 WMMAs, interleaved across 4 chains (each chain dependent only
      // every 4th issue): D = A*B + C
      c00 = __builtin_amdgcn_wmma_f32_16x16x32_bf16(false, a0h, false, b0h, (short)0, c00, false, false);
      c10 = __builtin_amdgcn_wmma_f32_16x16x32_bf16(false, a1h, false, b0h, (short)0, c10, false, false);
      c01 = __builtin_amdgcn_wmma_f32_16x16x32_bf16(false, a0h, false, b1h, (short)0, c01, false, false);
      c11 = __builtin_amdgcn_wmma_f32_16x16x32_bf16(false, a1h, false, b1h, (short)0, c11, false, false);
      c00 = __builtin_amdgcn_wmma_f32_16x16x32_bf16(false, a0h, false, b0l, (short)0, c00, false, false);
      c10 = __builtin_amdgcn_wmma_f32_16x16x32_bf16(false, a1h, false, b0l, (short)0, c10, false, false);
      c01 = __builtin_amdgcn_wmma_f32_16x16x32_bf16(false, a0h, false, b1l, (short)0, c01, false, false);
      c11 = __builtin_amdgcn_wmma_f32_16x16x32_bf16(false, a1h, false, b1l, (short)0, c11, false, false);
      c00 = __builtin_amdgcn_wmma_f32_16x16x32_bf16(false, a0l, false, b0h, (short)0, c00, false, false);
      c10 = __builtin_amdgcn_wmma_f32_16x16x32_bf16(false, a1l, false, b0h, (short)0, c10, false, false);
      c01 = __builtin_amdgcn_wmma_f32_16x16x32_bf16(false, a0l, false, b1h, (short)0, c01, false, false);
      c11 = __builtin_amdgcn_wmma_f32_16x16x32_bf16(false, a1l, false, b1h, (short)0, c11, false, false);
    }
    // Elementwise: C layout lane L, VGPR v -> (row = v + 8*(L>=16), col = eb*16 + L&15)
#pragma unroll
    for (int v = 0; v < 8; ++v) {
      int r = v + mhalf * 8;
      float d00 = lds_delta(aw, REG_A0H, REG_A0L, r, eb0 * 16 + n16);
      float d01 = lds_delta(aw, REG_A0H, REG_A0L, r, eb1 * 16 + n16);
      float d10 = lds_delta(aw, REG_A1H, REG_A1L, r, eb0 * 16 + n16);
      float d11 = lds_delta(aw, REG_A1H, REG_A1L, r, eb1 * 16 + n16);
      acc0[v] += c00[v] * d00 + c01[v] * d01;
      acc1[v] += c10[v] * d10 + c11[v] * d11;
    }
  }

  // ---- Reduce over the 16 lanes sharing each row; lanes 0 / 16 write out.
#pragma unroll
  for (int v = 0; v < 8; ++v) {
    float t0 = acc0[v];
    t0 += __shfl_xor(t0, 1);
    t0 += __shfl_xor(t0, 2);
    t0 += __shfl_xor(t0, 4);
    t0 += __shfl_xor(t0, 8);
    if ((lane & 15) == 0) out[rowBase + mhalf * 8 + v] = t0;
    float t1 = acc1[v];
    t1 += __shfl_xor(t1, 1);
    t1 += __shfl_xor(t1, 2);
    t1 += __shfl_xor(t1, 4);
    t1 += __shfl_xor(t1, 8);
    if ((lane & 15) == 0) out[rowBase + 16 + mhalf * 8 + v] = t1;
  }
}

extern "C" void kernel_launch(void* const* d_in, const int* in_sizes, int n_in,
                              void* d_out, int out_size, void* d_ws, size_t ws_size,
                              hipStream_t stream) {
  const float* x  = (const float*)d_in[0];
  const float* xf = (const float*)d_in[1];
  const float* S  = (const float*)d_in[2];
  float* out = (float*)d_out;

  // Workspace: bf16 hi/lo of S in B-operand layout (512 KB each).
  uint16_t* SBhi = (uint16_t*)d_ws;
  uint16_t* SBlo = (uint16_t*)((char*)d_ws + (size_t)DIM * DIM * sizeof(uint16_t));

  const int n = in_sizes[0] / DIM;  // 65536

  MahalanobisLayer_prepS<<<(DIM * DIM) / 512, 512, 0, stream>>>(S, SBhi, SBlo);
  MahalanobisLayer_main<<<n / 64, 64, 0, stream>>>(x, xf, SBhi, SBlo, out);
}
